// Decoder_16690242913225
// MI455X (gfx1250) — matrix operand order repaired and checked by reading, not compiled, not run on hardware
//
#include <hip/hip_runtime.h>

// ---------------------------------------------------------------------------
// Decoder: 64-step LSTM (teacher forcing) + 2x cross-attention + out proj +
// exact 1.5-entmax.  All GEMMs go through v_wmma_f32_16x16x32_f16 (wave32,
// 4 waves/block, 32x32 per wave).  B=512, T=64, E=512, V=256, Schar=64, Stag=16.
// ---------------------------------------------------------------------------

typedef _Float16 f16;
typedef __attribute__((ext_vector_type(16))) _Float16 v16h;
typedef __attribute__((ext_vector_type(8)))  _Float16 v8h;
typedef __attribute__((ext_vector_type(8)))  float    v8f;

union V16 { v16h v; v8h h[2]; };

__device__ __forceinline__ float sigmf_(float x) { return 1.f / (1.f + expf(-x)); }

// ---------------------------------------------------------------------------
// Unified WMMA GEMM:  D = alpha * (A (MxK,f16) * W(N,K,f16 row-major)^T + bias)
//   - split A operand (A0 for k<KA, A1 for k>=KA) -> LSTM [x|h] gates GEMM
//   - batched via blockIdx.z with per-operand batch strides (attention)
//   - MODE 0: f32 store   MODE 1: f16 store w/ row remap   MODE 2: f16 transposed
// Requires M%64==0, N%64==0, K%32==0 (all shapes here satisfy this).
// ---------------------------------------------------------------------------
template<int MODE>
__global__ __launch_bounds__(128) void gemm_k(
    const f16* __restrict__ A0, long lda0, long Abatch, int KA,
    const f16* __restrict__ A1, long lda1,
    const f16* __restrict__ Wm, long ldw, long Wbatch,
    const float* __restrict__ bias, float alpha,
    void* __restrict__ Dst, long ldo, long Dbatch, long colOff,
    int K, int rS, int rSp, long bStride)
{
  const int lane = threadIdx.x & 31;
  const int wv   = threadIdx.x >> 5;          // wave id 0..3
  const int wm   = wv >> 1, wn = wv & 1;      // 2x2 wave grid -> 64x64 block tile
  const long z   = blockIdx.z;
  const int m0 = blockIdx.y * 64 + wm * 32;
  const int n0 = blockIdx.x * 64 + wn * 32;
  const int lr = lane & 15;                   // row (A) / col (B) within 16-tile
  const int lh = lane >> 4;                   // lane half -> K half

  const f16* Ab = A0 + z * Abatch;
  const f16* Wb = Wm + z * Wbatch;

  v8f acc[2][2] = {};

  for (int k = 0; k < K; k += 32) {
    const f16* ap; long lda; int kk;
    if (k < KA) { ap = Ab; lda = lda0; kk = k; }
    else        { ap = A1; lda = lda1; kk = k - KA; }
    V16 av[2], bv[2];
#pragma unroll
    for (int i = 0; i < 2; i++) {
      // A (16x32 f16): lane m holds K {16h..16h+15}
      const f16* arow = ap + (long)(m0 + i * 16 + lr) * lda + kk;
      av[i].v = *(const v16h*)(arow + 16 * lh);
      // B (32x16): lane n holds K {16h..16h+15}; W is (N,K) row-major -> contiguous
      const f16* brow = Wb + (long)(n0 + i * 16 + lr) * ldw + k + 16 * lh;
      bv[i].v = *(const v16h*)brow;
      __builtin_prefetch(brow + 32, 0, 1);    // -> global_prefetch_b8
    }
#pragma unroll
    for (int i = 0; i < 2; i++)
#pragma unroll
      for (int j = 0; j < 2; j++)
        acc[i][j] = __builtin_amdgcn_wmma_f32_16x16x32_f16(
            false, av[i].v, false, bv[j].v, (short)0, acc[i][j], false, false);
  }

  // C/D layout: vgpr r -> M = r (lanes 0-15) / r+8 (lanes 16-31); N = lane&15
#pragma unroll
  for (int i = 0; i < 2; i++)
#pragma unroll
    for (int j = 0; j < 2; j++) {
      const int nn = n0 + j * 16 + lr;
      const float bvv = bias ? bias[nn] : 0.f;
#pragma unroll
      for (int r = 0; r < 8; r++) {
        const int mm = m0 + i * 16 + lh * 8 + r;
        const float val = alpha * (acc[i][j][r] + bvv);
        if (MODE == 0) {
          ((float*)Dst)[z * Dbatch + (long)mm * ldo + colOff + nn] = val;
        } else if (MODE == 1) {
          const long rr = (long)(mm / rS) * rSp + (mm % rS);
          ((f16*)Dst)[z * Dbatch + rr * ldo + colOff + nn] = (f16)val;
        } else {  // transposed per-batch store: vT[b][n][s]
          const long bb = mm / rS; const int ss = mm % rS;
          ((f16*)Dst)[bb * bStride + (long)nn * rSp + ss] = (f16)val;
        }
      }
    }
}

// --- LSTM elementwise cell (torch gate order i,f,g,o) ----------------------
__global__ void lstm_cell_k(const float* __restrict__ G, float* __restrict__ C,
                            f16* __restrict__ H16, f16* __restrict__ HSt)
{
  const int idx = blockIdx.x * blockDim.x + threadIdx.x;   // < 512*512
  const int b = idx >> 9, e = idx & 511;
  const float* g = G + (long)b * 2048;
  const float ig = g[e], fg = g[512 + e], gg = g[1024 + e], og = g[1536 + e];
  const float c = C[idx];
  const float cn = sigmf_(fg) * c + sigmf_(ig) * tanhf(gg);
  const float hn = sigmf_(og) * tanhf(cn);
  C[idx] = cn;
  const f16 h = (f16)hn;
  H16[idx] = h;
  HSt[idx] = h;
}

// --- f32 -> T strided convert/copy ----------------------------------------
template<typename T>
__global__ void convert_k(const float* __restrict__ src, T* __restrict__ dst,
                          long rows, long cols, long ldd, long colOff)
{
  const long idx = blockIdx.x * (long)blockDim.x + threadIdx.x;
  if (idx >= rows * cols) return;
  const long r = idx / cols, c = idx % cols;
  dst[r * ldd + colOff + c] = (T)src[idx];
}

// --- teacher-forcing inputs: xs[0]=0, xs[t]=true_out[:,t]; layout (t,b,v) f16
__global__ void build_xs_k(const float* __restrict__ tos, f16* __restrict__ xsb)
{
  const long idx = blockIdx.x * (long)blockDim.x + threadIdx.x;
  if (idx >= 64L * 512 * 256) return;
  const int t = (int)(idx / (512 * 256));
  const long r = idx % (512 * 256);
  const int b = (int)(r >> 8), v = (int)(r & 255);
  const float val = (t == 0) ? 0.f : tos[(long)b * 64 * 256 + (long)t * 256 + v];
  xsb[idx] = (f16)val;
}

__global__ void add_bias_k(const float* a, const float* b, float* o, int n)
{
  const int i = blockIdx.x * blockDim.x + threadIdx.x;
  if (i < n) o[i] = a[i] + b[i];
}

// --- softmax over 64 (char) -----------------------------------------------
__global__ __launch_bounds__(64) void softmax64_k(const float* __restrict__ S,
                                                  f16* __restrict__ P)
{
  __shared__ float sm[64];
  const long row = blockIdx.x; const int tid = threadIdx.x;
  const float v = S[row * 64 + tid];
  sm[tid] = v; __syncthreads();
  for (int o = 32; o > 0; o >>= 1) { if (tid < o) sm[tid] = fmaxf(sm[tid], sm[tid + o]); __syncthreads(); }
  const float mx = sm[0]; __syncthreads();
  const float e = expf(v - mx);
  sm[tid] = e; __syncthreads();
  for (int o = 32; o > 0; o >>= 1) { if (tid < o) sm[tid] += sm[tid + o]; __syncthreads(); }
  P[row * 64 + tid] = (f16)(e / sm[0]);
}

// --- softmax over 16 (tag), stored padded to 32 with zeros ----------------
__global__ __launch_bounds__(32) void softmax16_k(const float* __restrict__ S,
                                                  f16* __restrict__ P)
{
  __shared__ float sm[32];
  const long row = blockIdx.x; const int tid = threadIdx.x;
  const float v = (tid < 16) ? S[row * 64 + tid] : -3.0e38f;
  sm[tid] = v; __syncthreads();
  for (int o = 16; o > 0; o >>= 1) { if (tid < o) sm[tid] = fmaxf(sm[tid], sm[tid + o]); __syncthreads(); }
  const float mx = sm[0]; __syncthreads();
  const float e = (tid < 16) ? expf(v - mx) : 0.f;
  sm[tid] = e; __syncthreads();
  for (int o = 16; o > 0; o >>= 1) { if (tid < o) sm[tid] += sm[tid + o]; __syncthreads(); }
  P[row * 32 + tid] = (f16)(e / sm[0]);   // pad lanes write exact 0
}

// --- exact sort-based 1.5-entmax over V=256, write transposed (b,t,v) -----
__global__ __launch_bounds__(256) void entmax_k(const float* __restrict__ LG,
                                                float* __restrict__ out)
{
  __shared__ float sA[256], sB[256], sC[256];
  const long row = blockIdx.x;          // row = t*512 + b
  const int tid = threadIdx.x;
  const float z0 = 0.5f * LG[row * 256 + tid];
  sB[tid] = z0; __syncthreads();
  for (int o = 128; o > 0; o >>= 1) { if (tid < o) sB[tid] = fmaxf(sB[tid], sB[tid + o]); __syncthreads(); }
  const float zi = z0 - sB[0];
  __syncthreads();
  sA[tid] = zi; __syncthreads();
  // bitonic sort, descending
  for (int k2 = 2; k2 <= 256; k2 <<= 1)
    for (int j = k2 >> 1; j > 0; j >>= 1) {
      const int ixj = tid ^ j;
      if (ixj > tid) {
        const float a = sA[tid], c = sA[ixj];
        const bool desc = ((tid & k2) == 0);
        if ((a < c) == desc) { sA[tid] = c; sA[ixj] = a; }
      }
      __syncthreads();
    }
  const float zs = sA[tid];
  sB[tid] = zs; sC[tid] = zs * zs; __syncthreads();
  for (int o = 1; o < 256; o <<= 1) {               // inclusive scans
    const float x = (tid >= o) ? sB[tid - o] : 0.f;
    const float y = (tid >= o) ? sC[tid - o] : 0.f;
    __syncthreads();
    sB[tid] += x; sC[tid] += y;
    __syncthreads();
  }
  const float rho  = (float)(tid + 1);
  const float mean = sB[tid] / rho, msq = sC[tid] / rho;
  const float ssv  = rho * (msq - mean * mean);
  const float dlt  = (1.f - ssv) / rho;
  const float tau  = mean - ((dlt > 0.f) ? sqrtf(dlt) : 0.f);
  const int flag   = (tau <= zs) ? 1 : 0;
  __syncthreads();
  sB[tid] = tau; sC[tid] = (float)flag; __syncthreads();
  for (int o = 128; o > 0; o >>= 1) { if (tid < o) sC[tid] += sC[tid + o]; __syncthreads(); }
  const int kk = (int)sC[0];
  const float tstar = sB[kk - 1];
  const float y = fmaxf(zi - tstar, 0.f);
  const long t = row >> 9, b = row & 511;
  out[b * (64L * 256) + t * 256 + tid] = y * y;
}

// ---------------------------------------------------------------------------
extern "C" void kernel_launch(void* const* d_in, const int* in_sizes, int n_in,
                              void* d_out, int out_size, void* d_ws, size_t ws_size,
                              hipStream_t stream)
{
  (void)in_sizes; (void)n_in; (void)out_size; (void)ws_size;
  const float* char_enc = (const float*)d_in[0];
  const float* char_hn0 = (const float*)d_in[1];
  const float* char_cn0 = (const float*)d_in[2];
  const float* tag_enc  = (const float*)d_in[3];
  const float* tag_hn0  = (const float*)d_in[4];
  const float* tag_cn0  = (const float*)d_in[5];
  const float* tos      = (const float*)d_in[6];
  const float* w_ih     = (const float*)d_in[7];
  const float* w_hh     = (const float*)d_in[8];
  const float* b_ih     = (const float*)d_in[9];
  const float* b_hh     = (const float*)d_in[10];
  const float* wAttn[8] = { (const float*)d_in[11], (const float*)d_in[12],
                            (const float*)d_in[13], (const float*)d_in[17],
                            (const float*)d_in[19], (const float*)d_in[20],
                            (const float*)d_in[21], (const float*)d_in[25] };
  const float* char_bq = (const float*)d_in[14];
  const float* char_bk = (const float*)d_in[15];
  const float* char_bv = (const float*)d_in[16];
  const float* char_bo = (const float*)d_in[18];
  const float* tag_bq  = (const float*)d_in[22];
  const float* tag_bk  = (const float*)d_in[23];
  const float* tag_bv  = (const float*)d_in[24];
  const float* tag_bo  = (const float*)d_in[26];
  const float* out_w   = (const float*)d_in[27];
  const float* out_b   = (const float*)d_in[28];

  // ---- bump allocator over d_ws (~350 MB total) ----
  char* wsb = (char*)d_ws; size_t off = 0;
  auto alloc = [&](size_t bytes) -> char* {
    char* p = wsb + off; off = (off + bytes + 255) & ~(size_t)255; return p;
  };
  f16*   WG    = (f16*)  alloc(2048L * 768 * 2);        // [w_ih | w_hh] (N=2048,K=768)
  f16*   XSB   = (f16*)  alloc(64L * 512 * 256 * 2);    // teacher inputs (t,b,v)
  float* BSUM  = (float*)alloc(2048L * 4);
  float* GATES = (float*)alloc(512L * 2048 * 4);
  float* Cst   = (float*)alloc(512L * 512 * 4);
  f16*   H16   = (f16*)  alloc(512L * 512 * 2);
  f16*   HS    = (f16*)  alloc(32768L * 512 * 2);       // all hidden states (t,b,e)
  f16*   W8    = (f16*)  alloc(8L * 512 * 512 * 2);     // cq ck cv co tq tk tv to
  f16*   OUTW  = (f16*)  alloc(256L * 1024 * 2);
  f16*   CENC  = (f16*)  alloc(32768L * 512 * 2);
  f16*   TENC  = (f16*)  alloc(8192L * 512 * 2);
  f16*   Q     = (f16*)  alloc(32768L * 512 * 2);       // reused as O_char
  f16*   KC    = (f16*)  alloc(32768L * 512 * 2);       // reused as O_tag
  f16*   VTC   = (f16*)  alloc(512L * 512 * 64 * 2);    // vT char; reused as LOGITS f32
  f16*   KT    = (f16*)  alloc(512L * 64 * 512 * 2);    // tag K padded to 64 rows
  f16*   VTT   = (f16*)  alloc(512L * 512 * 32 * 2);    // vT tag, K padded to 32
  float* SC    = (float*)alloc(512L * 64 * 64 * 4);
  f16*   PC    = (f16*)  alloc(512L * 64 * 64 * 2);
  float* ST    = (float*)alloc(512L * 64 * 64 * 4);
  f16*   PT    = (f16*)  alloc(512L * 64 * 32 * 2);
  f16*   CAT   = (f16*)  alloc(32768L * 1024 * 2);
  f16*   OC = Q; f16* OT = KC; float* LOGITS = (float*)VTC;

  const int TPB = 256;
  auto nb = [](long n) { return (unsigned)((n + 255) / 256); };

  // ---- prep: conversions / packing ----
  convert_k<f16><<<nb(2048L*256), TPB, 0, stream>>>(w_ih, WG, 2048, 256, 768, 0);
  convert_k<f16><<<nb(2048L*512), TPB, 0, stream>>>(w_hh, WG, 2048, 512, 768, 256);
  add_bias_k<<<nb(2048), TPB, 0, stream>>>(b_ih, b_hh, BSUM, 2048);
  build_xs_k<<<nb(64L*512*256), TPB, 0, stream>>>(tos, XSB);
  convert_k<f16><<<nb(512L*256), TPB, 0, stream>>>(char_hn0, H16, 512, 256, 512, 0);
  convert_k<f16><<<nb(512L*256), TPB, 0, stream>>>(tag_hn0,  H16, 512, 256, 512, 256);
  convert_k<float><<<nb(512L*256), TPB, 0, stream>>>(char_cn0, Cst, 512, 256, 512, 0);
  convert_k<float><<<nb(512L*256), TPB, 0, stream>>>(tag_cn0,  Cst, 512, 256, 512, 256);
  convert_k<f16><<<nb(32768L*512), TPB, 0, stream>>>(char_enc, CENC, 32768, 512, 512, 0);
  convert_k<f16><<<nb(8192L*512),  TPB, 0, stream>>>(tag_enc,  TENC, 8192, 512, 512, 0);
  for (int i = 0; i < 8; i++)
    convert_k<f16><<<nb(512L*512), TPB, 0, stream>>>(wAttn[i], W8 + (long)i*512*512,
                                                     512, 512, 512, 0);
  convert_k<f16><<<nb(256L*1024), TPB, 0, stream>>>(out_w, OUTW, 256, 1024, 1024, 0);
  hipMemsetAsync(VTT, 0, 512L * 512 * 32 * 2, stream);  // zero K-pad (avoid NaN poison)

  f16 *WQc = W8, *WKc = W8 + 262144, *WVc = W8 + 2*262144, *WOc = W8 + 3*262144;
  f16 *WQt = W8 + 4*262144, *WKt = W8 + 5*262144, *WVt = W8 + 6*262144, *WOt = W8 + 7*262144;
  const int BIG = 1 << 28;

  // ---- LSTM: 64 sequential steps; gates = [x_t | h_{t-1}] @ [w_ih|w_hh]^T ----
  for (int t = 0; t < 64; ++t) {
    gemm_k<0><<<dim3(32, 8, 1), 128, 0, stream>>>(
        XSB + (long)t * 512 * 256, 256, 0, 256,   // A0 = x_t (512x256)
        H16, 512,                                  // A1 = h   (512x512)
        WG, 768, 0, BSUM, 1.f,
        GATES, 2048, 0, 0, 768, BIG, 0, 0);
    lstm_cell_k<<<1024, 256, 0, stream>>>(GATES, Cst, H16, HS + (long)t * 512 * 512);
  }

  // ---- K / V projections (V stored transposed per batch for AV GEMM) ----
  gemm_k<1><<<dim3(8, 512, 1), 128, 0, stream>>>(CENC, 512, 0, 512, nullptr, 0,
      WKc, 512, 0, char_bk, 1.f, KC, 512, 0, 0, 512, BIG, 0, 0);
  gemm_k<2><<<dim3(8, 512, 1), 128, 0, stream>>>(CENC, 512, 0, 512, nullptr, 0,
      WVc, 512, 0, char_bv, 1.f, VTC, 0, 0, 0, 512, 64, 64, 32768);
  gemm_k<1><<<dim3(8, 128, 1), 128, 0, stream>>>(TENC, 512, 0, 512, nullptr, 0,
      WKt, 512, 0, tag_bk, 1.f, KT, 512, 0, 0, 512, 16, 64, 0);       // pad 16->64 rows
  gemm_k<2><<<dim3(8, 128, 1), 128, 0, stream>>>(TENC, 512, 0, 512, nullptr, 0,
      WVt, 512, 0, tag_bv, 1.f, VTT, 0, 0, 0, 512, 16, 32, 16384);

  const float qscale = 0.04419417382415922f;  // 1/sqrt(512)

  // ---- char attention ----
  gemm_k<1><<<dim3(8, 512, 1), 128, 0, stream>>>(HS, 512, 0, 512, nullptr, 0,
      WQc, 512, 0, char_bq, qscale, Q, 512, 0, 0, 512, BIG, 0, 0);
  gemm_k<0><<<dim3(1, 1, 512), 128, 0, stream>>>(Q, 262144, 512, 512, nullptr, 0,
      KC, 512, 32768, nullptr, 1.f, SC, 64, 4096, 0, 512, 0, 0, 0);
  softmax64_k<<<32768, 64, 0, stream>>>(SC, PC);

  // ---- tag attention (Q buffer reused; char scores already consumed) ----
  gemm_k<1><<<dim3(8, 512, 1), 128, 0, stream>>>(HS, 512, 0, 512, nullptr, 0,
      WQt, 512, 0, tag_bq, qscale, Q, 512, 0, 0, 512, BIG, 0, 0);
  gemm_k<0><<<dim3(1, 1, 512), 128, 0, stream>>>(Q, 262144, 512, 512, nullptr, 0,
      KT, 512, 32768, nullptr, 1.f, ST, 64, 4096, 0, 512, 0, 0, 0);
  softmax16_k<<<32768, 32, 0, stream>>>(ST, PT);

  // ---- AV: O[t,b,e] = sum_s P[b,t,s] * V[b,s,e]  (W = vT, contiguous) ----
  gemm_k<1><<<dim3(8, 1, 512), 128, 0, stream>>>(PC, 64, 4096, 64, nullptr, 0,
      VTC, 64, 32768, nullptr, 1.f, OC, 262144, 512, 0, 64, BIG, 0, 0);
  gemm_k<1><<<dim3(8, 1, 512), 128, 0, stream>>>(PT, 32, 2048, 32, nullptr, 0,
      VTT, 32, 16384, nullptr, 1.f, OT, 262144, 512, 0, 32, BIG, 0, 0);

  // ---- output projections into concat buffer, then logits ----
  gemm_k<1><<<dim3(8, 512, 1), 128, 0, stream>>>(OC, 512, 0, 512, nullptr, 0,
      WOc, 512, 0, char_bo, 1.f, CAT, 1024, 0, 0, 512, BIG, 0, 0);
  gemm_k<1><<<dim3(8, 512, 1), 128, 0, stream>>>(OT, 512, 0, 512, nullptr, 0,
      WOt, 512, 0, tag_bo, 1.f, CAT, 1024, 0, 512, 512, BIG, 0, 0);
  gemm_k<0><<<dim3(4, 512, 1), 128, 0, stream>>>(CAT, 1024, 0, 1024, nullptr, 0,
      OUTW, 1024, 0, out_b, 1.f, LOGITS, 256, 0, 0, 1024, 0, 0, 0);

  // ---- 1.5-entmax + transpose to (B,T,V) ----
  entmax_k<<<32768, 256, 0, stream>>>(LOGITS, (float*)d_out);
}
